// PointTransformerLayer_37091337568769
// MI455X (gfx1250) — compile-verified, matrix operand order checked by reading, not run on hardware
//
#include <hip/hip_runtime.h>

typedef __attribute__((ext_vector_type(16))) _Float16 v16h;
typedef __attribute__((ext_vector_type(8)))  _Float16 v8h;
typedef __attribute__((ext_vector_type(8)))  float    v8f;
typedef __attribute__((ext_vector_type(4)))  int      v4i;

constexpr int B = 2, N = 8192, D = 128, K = 16;
constexpr int TILE_P = 16;
constexpr int ROWS = TILE_P * K;   // 256 neighbor rows per block
constexpr int LS = 136;            // padded row stride in halves (breaks bank conflicts)

// ---- optional CDNA5 async global->LDS path (guarded; falls back to plain copies)
#if defined(__has_builtin)
#if __has_builtin(__builtin_amdgcn_global_load_async_to_lds_b128)
#define HAVE_ASYNC_LDS 1
#endif
#endif
#ifndef HAVE_ASYNC_LDS
#define HAVE_ASYNC_LDS 0
#endif

#if HAVE_ASYNC_LDS
typedef __attribute__((address_space(1))) v4i* gas1_t;   // global v4i*
typedef __attribute__((address_space(3))) v4i* las3_t;   // LDS v4i* (32-bit)
static __device__ __forceinline__ void async_cp16(const void* g, void* l) {
  __builtin_amdgcn_global_load_async_to_lds_b128(
      (gas1_t)(unsigned long long)g,
      (las3_t)(unsigned int)(unsigned long long)l, 0, 0);
}
#endif

static __device__ __forceinline__ void wait_async_then_barrier() {
#if HAVE_ASYNC_LDS
#if __has_builtin(__builtin_amdgcn_s_wait_asynccnt)
  __builtin_amdgcn_s_wait_asynccnt(0);
#else
  asm volatile("s_wait_asynccnt 0x0" ::: "memory");
#endif
#endif
  __syncthreads();
}

// Copy one D x D f16 weight (row-major, stride D) into LDS with padded stride LS.
static __device__ __forceinline__
void stage_w(_Float16* s_w, const _Float16* W, int tid, int nthr) {
  for (int c = tid; c < D * (D / 8); c += nthr) {   // 2048 16-byte chunks
    const int row = c >> 4, col = (c & 15) * 8;
#if HAVE_ASYNC_LDS
    async_cp16(W + (size_t)row * D + col, &s_w[row * LS + col]);
#else
    *(v8h*)&s_w[row * LS + col] = *(const v8h*)&W[(size_t)row * D + col];
#endif
  }
}

union Frag { v16h v; v8h h[2]; };

// Load a 16x32 f16 A/B WMMA fragment from a row-major f16 buffer.
// Per ISA layout: lane<16 -> row=lane, K-groups {0..7, 16..23};
//                 lane>=16 -> row=lane-16, K-groups {8..15, 24..31}.
static __device__ __forceinline__
v16h load_frag(const _Float16* buf, int stride, int row0, int col0, int lane) {
  const int m = lane & 15, hi = lane >> 4;
  const _Float16* p = buf + (size_t)(row0 + m) * stride + col0 + hi * 8;
  Frag f;
  f.h[0] = *(const v8h*)p;        // ds_load_b128
  f.h[1] = *(const v8h*)(p + 16);
  return f.v;
}

static __device__ __forceinline__
v8f wmma_step(v16h a, v16h b, v8f c) {
  return __builtin_amdgcn_wmma_f32_16x16x32_f16(false, a, false, b, (short)0, c,
                                                false, false);
}

// One 256x128 GEMM: dst = act(src @ W^T + bias). Weight staged in LDS first.
// 8 waves x 2 row-tiles; A-fragments held in VGPRs across all 8 column tiles.
static __device__ __forceinline__
void gemm_all(const _Float16* src, _Float16* dst, _Float16* s_w, const _Float16* W,
              const float* __restrict__ bias, bool relu, int tid) {
  const int lane = tid & 31, wave = tid >> 5;
  stage_w(s_w, W, tid, 256);
  wait_async_then_barrier();
  for (int rt = wave * 2; rt < wave * 2 + 2; ++rt) {
    const int rb = rt * 16;
    const v16h a0 = load_frag(src, LS, rb, 0,  lane);
    const v16h a1 = load_frag(src, LS, rb, 32, lane);
    const v16h a2 = load_frag(src, LS, rb, 64, lane);
    const v16h a3 = load_frag(src, LS, rb, 96, lane);
    for (int ct = 0; ct < 8; ++ct) {
      const int oc = ct * 16;
      v8f acc = {};
      acc = wmma_step(a0, load_frag(s_w, LS, oc, 0,  lane), acc);
      acc = wmma_step(a1, load_frag(s_w, LS, oc, 32, lane), acc);
      acc = wmma_step(a2, load_frag(s_w, LS, oc, 64, lane), acc);
      acc = wmma_step(a3, load_frag(s_w, LS, oc, 96, lane), acc);
      const int col = oc + (lane & 15);
      const float bv = bias[col];
      const int rbase = rb + ((lane >> 4) << 3);
#pragma unroll
      for (int j = 0; j < 8; ++j) {
        float v = acc[j] + bv;
        if (relu) v = fmaxf(v, 0.f);
        dst[(rbase + j) * LS + col] = (_Float16)v;
      }
    }
  }
}

// ---------------- kernel 1: weights -> f16 ----------------
__global__ void convert_w_kernel(const float* __restrict__ a0, const float* __restrict__ a1,
                                 const float* __restrict__ a2, const float* __restrict__ a3,
                                 const float* __restrict__ a4, const float* __restrict__ a5,
                                 _Float16* __restrict__ out) {
  const int e = blockIdx.x * blockDim.x + threadIdx.x;
  if (e >= 6 * D * D) return;
  const int w = e >> 14, i = e & (D * D - 1);
  const float* s;
  switch (w) {
    case 0: s = a0; break; case 1: s = a1; break; case 2: s = a2; break;
    case 3: s = a3; break; case 4: s = a4; break; default: s = a5; break;
  }
  out[e] = (_Float16)s[i];
}

// ---------------- kernel 2: brute-force kNN (top-16) ----------------
__global__ void __launch_bounds__(256)
knn_kernel(const float* __restrict__ xyz, int* __restrict__ idxbuf) {
  __shared__ float s_xyz[256 * 3];
  const int tid = threadIdx.x;
  const int b = blockIdx.x / (N / 256);
  const int n = (blockIdx.x % (N / 256)) * 256 + tid;
  const float qx = xyz[((size_t)b * N + n) * 3 + 0];
  const float qy = xyz[((size_t)b * N + n) * 3 + 1];
  const float qz = xyz[((size_t)b * N + n) * 3 + 2];
  float best[K]; int bidx[K];
#pragma unroll
  for (int j = 0; j < K; ++j) { best[j] = 3.4e38f; bidx[j] = 0; }
  for (int c0 = 0; c0 < N; c0 += 256) {
    __syncthreads();
#pragma unroll
    for (int c = 0; c < 3; ++c)
      s_xyz[tid * 3 + c] = xyz[((size_t)b * N + c0 + tid) * 3 + c];
    __syncthreads();
    for (int cc = 0; cc < 256; ++cc) {
      const float dx = qx - s_xyz[cc * 3 + 0];
      const float dy = qy - s_xyz[cc * 3 + 1];
      const float dz = qz - s_xyz[cc * 3 + 2];
      const float d2 = dx * dx + dy * dy + dz * dz;
      if (d2 < best[K - 1]) {          // register-resident compare-swap chain
        float cd = d2; int ci = c0 + cc;
#pragma unroll
        for (int j = 0; j < K; ++j) {
          if (cd < best[j]) {
            float td = best[j]; int ti = bidx[j];
            best[j] = cd; bidx[j] = ci; cd = td; ci = ti;
          }
        }
      }
    }
  }
#pragma unroll
  for (int j = 0; j < K; ++j) idxbuf[((size_t)b * N + n) * K + j] = bidx[j];
}

// ---------------- kernel 3: q/k/v projections (per point, pre-gather) ----------------
__global__ void __launch_bounds__(128)
qkv_kernel(const float* __restrict__ feat, const _Float16* __restrict__ wf16,
           _Float16* __restrict__ qF, _Float16* __restrict__ kF, _Float16* __restrict__ vF) {
  extern __shared__ char smemq[];
  _Float16* s_x = (_Float16*)smemq;          // 16 x LS
  _Float16* s_w = s_x + TILE_P * LS;         // 3 x 128 x LS (w_q, w_k, w_v)
  const int tid = threadIdx.x, lane = tid & 31, wave = tid >> 5;
  const int b  = blockIdx.x / (N / TILE_P);
  const int n0 = (blockIdx.x % (N / TILE_P)) * TILE_P;

  // stage all three projection weights into LDS
  for (int c = tid; c < 3 * D * (D / 8); c += 128) {
    const int g = c >> 11, cc = c & 2047;
    const int row = cc >> 4, col = (cc & 15) * 8;
#if HAVE_ASYNC_LDS
    async_cp16(wf16 + (size_t)g * D * D + (size_t)row * D + col,
               &s_w[g * D * LS + row * LS + col]);
#else
    *(v8h*)&s_w[g * D * LS + row * LS + col] =
        *(const v8h*)&wf16[(size_t)g * D * D + (size_t)row * D + col];
#endif
  }
  for (int i = tid; i < TILE_P * D; i += 128)
    s_x[(i / D) * LS + (i % D)] =
        (_Float16)feat[((size_t)b * N + n0 + i / D) * D + (i % D)];
  wait_async_then_barrier();

  // A-fragments loaded once (single 16-row tile), reused by all 24 output tiles
  const v16h a0 = load_frag(s_x, LS, 0, 0,  lane);
  const v16h a1 = load_frag(s_x, LS, 0, 32, lane);
  const v16h a2 = load_frag(s_x, LS, 0, 64, lane);
  const v16h a3 = load_frag(s_x, LS, 0, 96, lane);
  for (int t = wave; t < 24; t += 4) {            // 3 gemms x 8 col-tiles
    const int g = t >> 3, oc = (t & 7) * 16;
    const _Float16* W = s_w + g * D * LS;
    v8f acc = {};
    acc = wmma_step(a0, load_frag(W, LS, oc, 0,  lane), acc);
    acc = wmma_step(a1, load_frag(W, LS, oc, 32, lane), acc);
    acc = wmma_step(a2, load_frag(W, LS, oc, 64, lane), acc);
    acc = wmma_step(a3, load_frag(W, LS, oc, 96, lane), acc);
    _Float16* dst = (g == 0) ? qF : (g == 1) ? kF : vF;
    const int col = oc + (lane & 15);
    const int rbase = (lane >> 4) << 3;
#pragma unroll
    for (int j = 0; j < 8; ++j)
      dst[((size_t)b * N + n0 + rbase + j) * D + col] = (_Float16)acc[j];
  }
}

// ---------------- kernel 4: fused attention + softmax + residual + LN ----------------
__global__ void __launch_bounds__(256)
attn_kernel(const float* __restrict__ xyz, const float* __restrict__ feat,
            const _Float16* __restrict__ qF, const _Float16* __restrict__ kF,
            const _Float16* __restrict__ vF, const int* __restrict__ idxbuf,
            const _Float16* __restrict__ wf16,
            const float* __restrict__ pe_w1, const float* __restrict__ pe_b1,
            const float* __restrict__ pe_b2, const float* __restrict__ am_b1,
            const float* __restrict__ am_b2, const float* __restrict__ ln_g,
            const float* __restrict__ ln_b, float* __restrict__ out) {
  extern __shared__ char smem[];
  _Float16* s_k  = (_Float16*)smem;            // 256xLS: k rows -> s=q-k+pe -> logits
  _Float16* s_pe = s_k + ROWS * LS;            // 256xLS: pos enc (kept to the end)
  _Float16* s_h  = s_pe + ROWS * LS;           // 256xLS: pe hidden -> am hidden
  _Float16* s_w  = s_h + ROWS * LS;            // 128xLS: staged weight (per phase)
  float*    s_q  = (float*)(s_w + D * LS);     // 16x128: q, later y
  float*    s_st = s_q + TILE_P * D;           // 16x2 LN stats
  int*      s_id = (int*)(s_st + 32);          // 256 neighbor indices

  const int tid = threadIdx.x;
  const int b  = blockIdx.x / (N / TILE_P);
  const int n0 = (blockIdx.x % (N / TILE_P)) * TILE_P;

  const _Float16* Wpe2 = wf16 + (size_t)3 * (D * D);
  const _Float16* Wam1 = wf16 + (size_t)4 * (D * D);
  const _Float16* Wam2 = wf16 + (size_t)5 * (D * D);

  // phase 0a: neighbor indices + q + pos-enc layer-1 params into LDS
  s_id[tid] = idxbuf[((size_t)b * N + n0 + (tid >> 4)) * K + (tid & 15)];
  float* s_pw = (float*)s_w;                    // reuse weight slot: 384 + 128 floats
  for (int i = tid; i < 512; i += 256)
    s_pw[i] = (i < 384) ? pe_w1[i] : pe_b1[i - 384];
  for (int i = tid; i < TILE_P * D; i += 256)
    s_q[i] = (float)qF[((size_t)b * N + n0 + i / D) * D + (i % D)];
  __syncthreads();

  // phase 0b: gather k rows (L2-resident) + pos-enc hidden layer (K-dim=3, VALU)
  {
    const int r = tid, p = r >> 4;
    const int nb = s_id[r];
    const _Float16* krow = kF + ((size_t)b * N + nb) * D;
#pragma unroll
    for (int d = 0; d < D; d += 8)
      *(v8h*)&s_k[r * LS + d] = *(const v8h*)&krow[d];
    const float dx = xyz[((size_t)b * N + n0 + p) * 3 + 0] - xyz[((size_t)b * N + nb) * 3 + 0];
    const float dy = xyz[((size_t)b * N + n0 + p) * 3 + 1] - xyz[((size_t)b * N + nb) * 3 + 1];
    const float dz = xyz[((size_t)b * N + n0 + p) * 3 + 2] - xyz[((size_t)b * N + nb) * 3 + 2];
    for (int o = 0; o < D; ++o) {
      float h = s_pw[384 + o] + s_pw[o * 3 + 0] * dx + s_pw[o * 3 + 1] * dy + s_pw[o * 3 + 2] * dz;
      s_h[r * LS + o] = (_Float16)fmaxf(h, 0.f);
    }
  }
  __syncthreads();

  // phase 1: pe = h1 @ pe_w2^T + b2  (WMMA)
  gemm_all(s_h, s_pe, s_w, Wpe2, pe_b2, false, tid);
  __syncthreads();

  // phase 2: s = q - k + pe  (in place into s_k)
  {
    const int r = tid, p = r >> 4;
    for (int d = 0; d < D; ++d)
      s_k[r * LS + d] =
          (_Float16)(s_q[p * D + d] - (float)s_k[r * LS + d] + (float)s_pe[r * LS + d]);
  }
  __syncthreads();

  // phase 3: h = relu(s @ am_w1^T + b1)  (WMMA)
  gemm_all(s_k, s_h, s_w, Wam1, am_b1, true, tid);
  __syncthreads();

  // phase 4: logits = h @ am_w2^T + b2  (WMMA, into s_k)
  gemm_all(s_h, s_k, s_w, Wam2, am_b2, false, tid);
  __syncthreads();

  // phase 5: per-channel softmax over K neighbors + weighted (v+pe) sum + residual
  {
    const int p = tid >> 4, d0 = (tid & 15) * 8;
    float mx[8];
#pragma unroll
    for (int j = 0; j < 8; ++j) mx[j] = -3.4e38f;
    for (int kk = 0; kk < K; ++kk) {
      const int rr = (p * K + kk) * LS + d0;
#pragma unroll
      for (int j = 0; j < 8; ++j) mx[j] = fmaxf(mx[j], (float)s_k[rr + j]);
    }
    float num[8] = {}, den[8] = {};
    for (int kk = 0; kk < K; ++kk) {
      const int rr = (p * K + kk) * LS + d0;
      const int nb = s_id[p * K + kk];
      const v8h vv = *(const v8h*)(vF + ((size_t)b * N + nb) * D + d0);
#pragma unroll
      for (int j = 0; j < 8; ++j) {
        const float e = __expf((float)s_k[rr + j] - mx[j]);
        den[j] += e;
        num[j] += e * ((float)vv[j] + (float)s_pe[rr + j]);
      }
    }
#pragma unroll
    for (int j = 0; j < 8; ++j)
      s_q[p * D + d0 + j] =
          num[j] / den[j] + feat[((size_t)b * N + n0 + p) * D + d0 + j];
  }
  __syncthreads();

  // phase 6: LayerNorm
  if (tid < TILE_P) {
    float s = 0.f, ss = 0.f;
    for (int d = 0; d < D; ++d) { const float y = s_q[tid * D + d]; s += y; ss += y * y; }
    const float mu = s * (1.0f / D);
    const float var = ss * (1.0f / D) - mu * mu;
    s_st[tid * 2 + 0] = mu;
    s_st[tid * 2 + 1] = rsqrtf(var + 1e-5f);
  }
  __syncthreads();
  for (int i = tid; i < TILE_P * D; i += 256) {
    const int p = i / D, d = i % D;
    out[((size_t)b * N + n0 + p) * D + d] =
        (s_q[i] - s_st[p * 2]) * s_st[p * 2 + 1] * ln_g[d] + ln_b[d];
  }
}

extern "C" void kernel_launch(void* const* d_in, const int* in_sizes, int n_in,
                              void* d_out, int out_size, void* d_ws, size_t ws_size,
                              hipStream_t stream) {
  const float* xyz   = (const float*)d_in[0];
  const float* feat  = (const float*)d_in[1];
  const float* w_q   = (const float*)d_in[2];
  const float* w_k   = (const float*)d_in[3];
  const float* w_v   = (const float*)d_in[4];
  const float* pe_w1 = (const float*)d_in[5];
  const float* pe_b1 = (const float*)d_in[6];
  const float* pe_w2 = (const float*)d_in[7];
  const float* pe_b2 = (const float*)d_in[8];
  const float* am_w1 = (const float*)d_in[9];
  const float* am_b1 = (const float*)d_in[10];
  const float* am_w2 = (const float*)d_in[11];
  const float* am_b2 = (const float*)d_in[12];
  const float* ln_g  = (const float*)d_in[13];
  const float* ln_b  = (const float*)d_in[14];
  float* out = (float*)d_out;

  char* ws = (char*)d_ws;
  _Float16* wf16 = (_Float16*)ws;                                    // 6*D*D halves
  int* idx = (int*)(ws + (size_t)6 * D * D * 2);                     // B*N*K ints
  _Float16* qF = (_Float16*)(ws + (size_t)6 * D * D * 2 + (size_t)B * N * K * 4);
  _Float16* kF = qF + (size_t)B * N * D;
  _Float16* vF = kF + (size_t)B * N * D;

  convert_w_kernel<<<(6 * D * D + 255) / 256, 256, 0, stream>>>(
      w_q, w_k, w_v, pe_w2, am_w1, am_w2, wf16);
  knn_kernel<<<B * (N / 256), 256, 0, stream>>>(xyz, idx);
  const size_t smem_qkv = (size_t)(TILE_P * LS + 3 * D * LS) * 2;
  qkv_kernel<<<B * (N / TILE_P), 128, smem_qkv, stream>>>(feat, wf16, qF, kF, vF);
  const size_t smem_attn = (size_t)3 * ROWS * LS * 2 + (size_t)D * LS * 2 +
                           (size_t)TILE_P * D * 4 + 32 * 4 + 256 * 4;
  attn_kernel<<<B * (N / TILE_P), 256, smem_attn, stream>>>(
      xyz, feat, qF, kF, vF, idx, wf16,
      pe_w1, pe_b1, pe_b2, am_b1, am_b2, ln_g, ln_b, out);
}
